// SepLSTMs_24412594110601
// MI455X (gfx1250) — matrix operand order repaired and checked by reading, not compiled, not run on hardware
//
#include <hip/hip_runtime.h>
#include <hip/hip_bf16.h>
#include <math.h>

#define B_  64
#define T_  256
#define F_  64
#define H_  512
#define G4H 2048
#define NT_ 128   // N tiles of 16 over 4H
#define KT_ 16    // K steps of 32 over H

typedef __attribute__((ext_vector_type(16))) __bf16 v16bf;
typedef __attribute__((ext_vector_type(8)))  float  v8f;

union Frag { v16bf v; uint4 q[2]; };

__device__ __forceinline__ unsigned short f2bf(float x) {
  unsigned u = __float_as_uint(x);
  return (unsigned short)((u + 0x7FFFu + ((u >> 16) & 1u)) >> 16);
}

// branch-free fast sigmoid/tanh: v_exp_f32 + v_rcp_f32, correct saturation at +-inf
__device__ __forceinline__ float sig_(float x) {
  return __builtin_amdgcn_rcpf(1.0f + __expf(-x));
}
__device__ __forceinline__ float tanh_(float x) {
  return 1.0f - 2.0f * __builtin_amdgcn_rcpf(1.0f + __expf(2.0f * x));
}

// ---------------------------------------------------------------------------
// Prepass: pack W_hh[f] (fp32 [4H,H]) into bf16 WMMA B-fragments.
// B matrix is K x N = H x 4H with B[k][n] = W_hh[n][k].
// Block (f, nt, kt) = 32x16 bf16 tile, stored lane-major (32 lanes x 8 dwords):
//   lane<16 : n = nt*16+lane,      K = kt*32 + {2j, 2j+1}
//   lane>=16: n = nt*16+(lane-16), K = kt*32 + 16 + {2j, 2j+1}
// ---------------------------------------------------------------------------
__global__ __launch_bounds__(256) void pack_whh(const float* __restrict__ W_hh,
                                                unsigned int* __restrict__ wsB) {
  unsigned tid  = blockIdx.x * 256u + threadIdx.x;  // 64*128*16*32 = 2^22 threads
  unsigned lane = tid & 31u;
  unsigned kt   = (tid >> 5) & 15u;
  unsigned nt   = (tid >> 9) & 127u;
  unsigned f    = tid >> 16;
  unsigned n      = nt * 16u + (lane & 15u);
  unsigned kstart = kt * 32u + ((lane < 16u) ? 0u : 16u);
  const float* src = W_hh + (size_t)f * G4H * H_ + (size_t)n * H_ + kstart;
  unsigned int o[8];
#pragma unroll
  for (int j = 0; j < 8; ++j) {
    unsigned lo = f2bf(src[2 * j]);
    unsigned hi = f2bf(src[2 * j + 1]);
    o[j] = lo | (hi << 16);
  }
  unsigned int* dst = wsB + (((size_t)f * NT_ + nt) * KT_ + kt) * 256u + lane * 8u;
  uint4* d4 = reinterpret_cast<uint4*>(dst);
  d4[0] = make_uint4(o[0], o[1], o[2], o[3]);
  d4[1] = make_uint4(o[4], o[5], o[6], o[7]);
}

// ---------------------------------------------------------------------------
// Main recurrence: 1 workgroup per feature, 1024 threads = 32 waves.
// wave = mt(0..3 batch tile) x grp(0..7 hidden group of 64 hidden columns).
// j-outer structure: for each hh-tile j (0..3), compute the 4 gate tiles
// (acc[4] = 32 VGPRs live) over the full K loop, then immediately apply the
// cell update -- accumulators never persist across j, so no spilling.
// h is double-buffered bf16 in LDS (read t&1, write (t&1)^1): one barrier/step.
// ---------------------------------------------------------------------------
__global__ __launch_bounds__(1024) void lstm_main(
    const float* __restrict__ feat, const float* __restrict__ mask,
    const float* __restrict__ h0,   const float* __restrict__ c0,
    const float* __restrict__ W_ih, const float* __restrict__ b_ih,
    const float* __restrict__ b_hh, const unsigned int* __restrict__ wsB,
    float* __restrict__ out) {
  __shared__ unsigned short h_lds[2][B_ * H_];  // 2 x 64 KB bf16 hidden state

  const int f    = blockIdx.x;
  const int tid  = threadIdx.x;
  const int wave = tid >> 5;
  const int lane = tid & 31;
  const int mt   = wave & 3;
  const int grp  = wave >> 2;
  const int nl   = lane & 15;
  const int hiH  = (lane >= 16) ? 8 : 0;

  // ---- h0 -> bf16 LDS buffer 0
  for (int i = 0; i < (B_ * H_) / 1024; ++i) {
    int e = tid + i * 1024;
    h_lds[0][e] = f2bf(h0[(size_t)f * B_ * H_ + e]);
  }

  // ---- c0 -> registers (this wave's private (b,hh) partition)
  float creg[4][8];
#pragma unroll
  for (int j = 0; j < 4; ++j)
#pragma unroll
    for (int r = 0; r < 8; ++r) {
      int b  = mt * 16 + r + hiH;
      int hh = (grp * 4 + j) * 16 + nl;
      creg[j][r] = c0[(size_t)f * B_ * H_ + (size_t)b * H_ + hh];
    }

  __syncthreads();

  const unsigned int* wsF  = wsB + (size_t)f * NT_ * KT_ * 256;
  const float*        wih  = W_ih + (size_t)f * G4H * 2;
  const float*        bi   = b_ih + (size_t)f * G4H;
  const float*        bh   = b_hh + (size_t)f * G4H;
  const int           arow = mt * 16 + nl;
  const int           klo  = (lane < 16) ? 0 : 16;

#pragma unroll 1
  for (int t = 0; t < T_; ++t) {
    const unsigned short* rb = h_lds[t & 1];
    unsigned short*       wb = h_lds[(t & 1) ^ 1];
    const bool last = (t == T_ - 1);

    // x_t for this wave's 8 batch rows (same for all j; L1/L2-resident)
    float xf[8], xm[8];
#pragma unroll
    for (int r = 0; r < 8; ++r) {
      int b = mt * 16 + r + hiH;
      size_t xi = (size_t)b * T_ * F_ + (size_t)t * F_ + f;
      xf[r] = feat[xi];
      xm[r] = mask[xi];
    }

#pragma unroll
    for (int j = 0; j < 4; ++j) {
      const int hh = (grp * 4 + j) * 16 + nl;

      // per-(j) gate-column coefficients (transient registers)
      float w0c[4], w1c[4], bbc[4];
#pragma unroll
      for (int g = 0; g < 4; ++g) {
        int n = (g * 32 + grp * 4 + j) * 16 + nl;
        w0c[g] = wih[n * 2 + 0];
        w1c[g] = wih[n * 2 + 1];
        bbc[g] = bi[n] + bh[n];
      }

      // init accumulators with x @ W_ih^T + bias
      v8f acc[4];
#pragma unroll
      for (int g = 0; g < 4; ++g)
#pragma unroll
        for (int r = 0; r < 8; ++r)
          acc[g][r] = xf[r] * w0c[g] + xm[r] * w1c[g] + bbc[g];

      // K loop: h @ W_hh^T via bf16 WMMA (only acc[4] = 32 VGPRs live)
#pragma unroll 1
      for (int kt = 0; kt < KT_; ++kt) {
        Frag a;
        int koff = kt * 32 + klo;
        a.q[0] = *reinterpret_cast<const uint4*>(&rb[arow * H_ + koff]);
        a.q[1] = *reinterpret_cast<const uint4*>(&rb[arow * H_ + koff + 8]);
        const unsigned int* bbase =
            wsF + ((size_t)(grp * 4 + j) * KT_ + kt) * 256 + lane * 8;
#pragma unroll
        for (int g = 0; g < 4; ++g) {
          Frag bfr;
          const uint4* bp =
              reinterpret_cast<const uint4*>(bbase + (size_t)g * 32 * KT_ * 256);
          bfr.q[0] = bp[0];
          bfr.q[1] = bp[1];
          acc[g] = __builtin_amdgcn_wmma_f32_16x16x32_bf16(
              false, a.v, false, bfr.v, (short)0, acc[g], false, false);
        }
      }

      // cell update for this hh tile; write h_new to the OTHER LDS buffer
#pragma unroll
      for (int r = 0; r < 8; ++r) {
        int b = mt * 16 + r + hiH;
        float ig = sig_(acc[0][r]);
        float fg = sig_(acc[1][r]);
        float gg = tanh_(acc[2][r]);
        float og = sig_(acc[3][r]);
        float cn = fg * creg[j][r] + ig * gg;
        creg[j][r] = cn;
        float hn = og * tanh_(cn);
        wb[b * H_ + hh] = f2bf(hn);
        if (last) out[(size_t)f * B_ * H_ + (size_t)b * H_ + hh] = hn;
      }
    }

    __syncthreads();  // all waves done: reads of rb complete, wb fully written
  }
}

extern "C" void kernel_launch(void* const* d_in, const int* in_sizes, int n_in,
                              void* d_out, int out_size, void* d_ws, size_t ws_size,
                              hipStream_t stream) {
  const float* feat = (const float*)d_in[0];
  const float* mask = (const float*)d_in[1];
  const float* h0   = (const float*)d_in[2];
  const float* c0   = (const float*)d_in[3];
  const float* W_ih = (const float*)d_in[4];
  const float* W_hh = (const float*)d_in[5];
  const float* b_ih = (const float*)d_in[6];
  const float* b_hh = (const float*)d_in[7];
  unsigned int* wsB = (unsigned int*)d_ws;  // 64*128*16*1024 B = 128 MB packed bf16 W_hh

  // prepass: 2^22 threads, one per 32-byte fragment slice
  pack_whh<<<16384, 256, 0, stream>>>(W_hh, wsB);
  // recurrence: one workgroup per feature
  lstm_main<<<F_, 1024, 0, stream>>>(feat, mask, h0, c0, W_ih, b_ih, b_hh, wsB,
                                     (float*)d_out);
}
